// MAD_72679436582977
// MI455X (gfx1250) — matrix operand order repaired and checked by reading, not compiled, not run on hardware
//
#include <hip/hip_runtime.h>

// ---------------------------------------------------------------------------
// Problem constants (from the reference)
// ---------------------------------------------------------------------------
#define N_HEADS   4
#define N_NODES   50000
#define DIM       32
#define N_BATCH   1024
#define N_QUERY   2048          // src(1024) + tgt(1024); flat edges[q] is the node id
#define N_NEAR    8
#define N_TILES   ((N_NODES + 15) / 16)   // 3125 node tiles of 16
#define N_PAIRS   ((N_TILES + 1) / 2)     // 1563 pairs of node tiles

typedef _Float16 v16h __attribute__((ext_vector_type(16)));
typedef _Float16 v8h  __attribute__((ext_vector_type(8)));
typedef float    v8f  __attribute__((ext_vector_type(8)));

// Workspace layout (bytes)
#define WS_EF16_OFF  0u                      // f16 embeds: 4*50000*32*2 = 12,800,000
#define WS_EN_OFF    12800000u               // f32 norms:  4*50000*4    =    800,000
#define WS_QF16_OFF  13600000u               // f16 query:  4*2048*32*2  =    524,288
#define WS_NBR_OFF   14124288u               // i32 nbrs:   4*2048*8*4   =    262,144

// ---------------------------------------------------------------------------
// K0: convert embeds f32 -> f16 and compute exact f32 squared node norms
// ---------------------------------------------------------------------------
__global__ __launch_bounds__(256) void MAD_cvt_norm_kernel(
    const float* __restrict__ embeds, _Float16* __restrict__ Ef16,
    float* __restrict__ En, int total_rows) {
  int row = blockIdx.x * blockDim.x + threadIdx.x;
  if (row >= total_rows) return;
  const float* src = embeds + (size_t)row * DIM;
  _Float16*    dst = Ef16   + (size_t)row * DIM;
  float nrm = 0.f;
#pragma unroll
  for (int j = 0; j < DIM; ++j) {
    float v = src[j];
    nrm += v * v;
    dst[j] = (_Float16)v;
  }
  En[row] = nrm;
}

// ---------------------------------------------------------------------------
// K0b: gather the 2048 query rows (per head) into a dense f16 matrix
// ---------------------------------------------------------------------------
__global__ __launch_bounds__(256) void MAD_gatherq_kernel(
    const float* __restrict__ embeds, const long long* __restrict__ edges,
    _Float16* __restrict__ Qf16) {
  int t = blockIdx.x * blockDim.x + threadIdx.x;   // t = h*2048 + q
  if (t >= N_HEADS * N_QUERY) return;
  int h = t >> 11;
  int q = t & (N_QUERY - 1);
  long long node = edges[q];                       // flat (2,1024): src then tgt
  const float* src = embeds + ((size_t)h * N_NODES + (size_t)node) * DIM;
  _Float16*    dst = Qf16 + (size_t)t * DIM;
#pragma unroll
  for (int j = 0; j < DIM; ++j) dst[j] = (_Float16)src[j];
}

// ---------------------------------------------------------------------------
// Sorted top-9 bubble insert (registers only, constant indices)
// ---------------------------------------------------------------------------
__device__ __forceinline__ void top9_insert(float (&best)[9], int (&bidx)[9],
                                            float s, int id) {
  float cs = s;
  int   ci = id;
#pragma unroll
  for (int i = 0; i < 9; ++i) {
    bool lt = cs < best[i];
    float tb = best[i]; int ti = bidx[i];
    if (lt) { best[i] = cs; bidx[i] = ci; cs = tb; ci = ti; }
  }
}

// ---------------------------------------------------------------------------
// K1: WMMA distance tiles + fused top-9 selection.
// One block per (head, 16-query tile). 8 waves stride over PAIRS of node
// tiles (2 WMMAs per loop body). Rank key: en[n] - 2*(q . e[n]).
// ---------------------------------------------------------------------------
__global__ __launch_bounds__(256) void MAD_knn_kernel(
    const _Float16* __restrict__ Ef16, const float* __restrict__ En,
    const _Float16* __restrict__ Qf16, const long long* __restrict__ edges,
    int* __restrict__ nbr) {
  __shared__ float scoreTile[8][2][16][16];   // per-wave two 16x16 score tiles
  __shared__ float mergeS[8][16][9];
  __shared__ int   mergeI[8][16][9];

  const int tid  = threadIdx.x;
  const int wave = tid >> 5;
  const int lane = tid & 31;
  const int col  = lane & 15;     // A: row M; B: column N; C: column N
  const int hsel = lane >> 4;

  const int h  = blockIdx.x >> 7;        // 4 heads x 128 query tiles
  const int qt = blockIdx.x & 127;

  // ---- Load A fragment: 16 queries x 32 dims (f16), contiguous k order (k = 16*hsel + elem) ----------
  // lane<16: elems 0..15 = K0..15
  // lane>=16: elems 0..15 = K16..31
  const _Float16* qrow =
      Qf16 + ((size_t)h * N_QUERY + (size_t)qt * 16 + col) * DIM;
  v8h alo = *(const v8h*)(qrow + 16 * hsel);
  v8h ahi = *(const v8h*)(qrow + 16 * hsel + 8);
  v16h afrag;
#pragma unroll
  for (int i = 0; i < 8; ++i) { afrag[i] = alo[i]; afrag[8 + i] = ahi[i]; }

  // ---- Private top-9 (ascending) per query row, lanes 0..15 own row=lane --
  float best[9];
  int   bidx[9];
#pragma unroll
  for (int i = 0; i < 9; ++i) { best[i] = 3.0e38f; bidx[i] = -1; }

  const _Float16* Eh  = Ef16 + (size_t)h * N_NODES * DIM;
  const float*    Enh = En + (size_t)h * N_NODES;

  for (int p = wave; p < N_PAIRS; p += 8) {
    const int  nodeBase0 = (2 * p) * 16;
    const bool twoTiles  = (2 * p + 1) < N_TILES;   // wave-uniform

    // ---- Tile 0: B fragment (contiguous k order, same as A: lane<16 K0..15, lane>=16 K16..31)
    {
      const _Float16* erow =
          Eh + ((size_t)(nodeBase0 + col)) * DIM + 16 * hsel;
      v8h blo = *(const v8h*)(erow);
      v8h bhi = *(const v8h*)(erow + 8);
      v16h bfrag;
#pragma unroll
      for (int i = 0; i < 8; ++i) { bfrag[i] = blo[i]; bfrag[8 + i] = bhi[i]; }

      v8f c = {};
      c = __builtin_amdgcn_wmma_f32_16x16x32_f16(
          false, afrag, false, bfrag, (short)0, c, false, false);

      const float en_n = Enh[nodeBase0 + col];   // norm of column node N=col
#pragma unroll
      for (int j = 0; j < 8; ++j)                // C: M = j + 8*hsel, N = col
        scoreTile[wave][0][j + 8 * hsel][col] = en_n - 2.0f * c[j];
    }

    // ---- Tile 1 (skipped only on the final odd pair; wave-uniform) --------
    if (twoTiles) {
      const int nodeBase1 = nodeBase0 + 16;
      const _Float16* erow =
          Eh + ((size_t)(nodeBase1 + col)) * DIM + 16 * hsel;
      v8h blo = *(const v8h*)(erow);
      v8h bhi = *(const v8h*)(erow + 8);
      v16h bfrag;
#pragma unroll
      for (int i = 0; i < 8; ++i) { bfrag[i] = blo[i]; bfrag[8 + i] = bhi[i]; }

      v8f c = {};
      c = __builtin_amdgcn_wmma_f32_16x16x32_f16(
          false, afrag, false, bfrag, (short)0, c, false, false);

      const float en_n = Enh[nodeBase1 + col];
#pragma unroll
      for (int j = 0; j < 8; ++j)
        scoreTile[wave][1][j + 8 * hsel][col] = en_n - 2.0f * c[j];
    }

    // wave-internal LDS ordering (DS ops are in-order per wave; fence the
    // compiler and make the cross-lane RAW explicit)
    asm volatile("s_wait_dscnt 0" ::: "memory");

    // ---- Selection: lane m owns query row m; vectorized row fetch ---------
    if (lane < 16) {
      const int m = lane;
#pragma unroll
      for (int t = 0; t < 2; ++t) {
        if (t == 1 && !twoTiles) break;
        const int nb = nodeBase0 + t * 16;
        const float4* rp = (const float4*)&scoreTile[wave][t][m][0];
        float4 r0 = rp[0], r1 = rp[1], r2 = rp[2], r3 = rp[3];
        float s[16] = {r0.x, r0.y, r0.z, r0.w, r1.x, r1.y, r1.z, r1.w,
                       r2.x, r2.y, r2.z, r2.w, r3.x, r3.y, r3.z, r3.w};
#pragma unroll
        for (int n = 0; n < 16; ++n) {
          if (s[n] < best[8]) top9_insert(best, bidx, s[n], nb + n);
        }
      }
    }
  }

  // ---- Per-wave results -> LDS, then 8-way merge by wave 0 ----------------
  if (lane < 16) {
#pragma unroll
    for (int k = 0; k < 9; ++k) {
      mergeS[wave][lane][k] = best[k];
      mergeI[wave][lane][k] = bidx[k];
    }
  }
  __syncthreads();

  if (wave == 0 && lane < 16) {
    const int m = lane;
    const int q = qt * 16 + m;                // global query id (0..2047)
    const int selfNode = (int)edges[q];

    float fb[9]; int fi[9];
#pragma unroll
    for (int i = 0; i < 9; ++i) { fb[i] = 3.0e38f; fi[i] = -1; }

    for (int w = 0; w < 8; ++w) {
#pragma unroll
      for (int k = 0; k < 9; ++k) {
        float s = mergeS[w][m][k];
        int   id = mergeI[w][m][k];
        if (id >= 0 && s < fb[8]) top9_insert(fb, fi, s, id);
      }
    }

    // emit first 8 non-self neighbors (ascending distance)
    int* dst = nbr + ((size_t)h * N_QUERY + q) * N_NEAR;
    int outk = 0;
#pragma unroll
    for (int i = 0; i < 9; ++i) {
      int id = fi[i];
      if (outk < N_NEAR && id >= 0 && id != selfNode) dst[outk++] = id;
    }
  }
}

// ---------------------------------------------------------------------------
// K2: gather neighbors, logits/dists in f32, softmax(24), head mean, sigmoid
// ---------------------------------------------------------------------------
__global__ __launch_bounds__(256) void MAD_final_kernel(
    const float* __restrict__ embeds, const float* __restrict__ field,
    const long long* __restrict__ edges, const int* __restrict__ nbr,
    float* __restrict__ out) {
  int b = blockIdx.x * blockDim.x + threadIdx.x;
  if (b >= N_BATCH) return;

  float pred = 0.f;
#pragma unroll 1
  for (int h = 0; h < N_HEADS; ++h) {
    float logits[16], dist[16];
#pragma unroll 1
    for (int side = 0; side < 2; ++side) {
      const int q = side * N_BATCH + b;
      const long long node = edges[q];
      const float* ep = embeds + ((size_t)h * N_NODES + (size_t)node) * DIM;
      const float* fp = field  + ((size_t)h * N_NODES + (size_t)node) * DIM;
      float e[DIM], f[DIM];
#pragma unroll
      for (int j = 0; j < DIM; ++j) { e[j] = ep[j]; f[j] = fp[j]; }
#pragma unroll 1
      for (int k = 0; k < N_NEAR; ++k) {
        const int n = nbr[((size_t)h * N_QUERY + q) * N_NEAR + k];
        const float* np = embeds + ((size_t)h * N_NODES + (size_t)n) * DIM;
        float lg = 0.f, d2 = 0.f;
#pragma unroll
        for (int j = 0; j < DIM; ++j) {
          float df = e[j] - np[j];
          lg += df * f[j];
          d2 += df * df;
        }
        logits[side * 8 + k] = lg;
        dist[side * 8 + k]   = sqrtf(d2);
      }
    }
    // softmax over [16 neighbor scores (1-dist), 8 sentinels (score 0)]
    float mx = 0.f;                         // sentinel score = 1 - 1 = 0
#pragma unroll
    for (int i = 0; i < 16; ++i) mx = fmaxf(mx, 1.f - dist[i]);
    float den = 8.f * __expf(0.f - mx);     // 8 sentinels, logit 0
    float num = 0.f;
#pragma unroll
    for (int i = 0; i < 16; ++i) {
      float w = __expf((1.f - dist[i]) - mx);
      den += w;
      num += logits[i] * w;
    }
    pred += num / den;
  }
  pred *= 0.25f;                            // mean over heads
  out[b] = 1.f / (1.f + __expf(-pred));
}

// ---------------------------------------------------------------------------
extern "C" void kernel_launch(void* const* d_in, const int* in_sizes, int n_in,
                              void* d_out, int out_size, void* d_ws,
                              size_t ws_size, hipStream_t stream) {
  const float*     embeds = (const float*)d_in[0];
  const float*     field  = (const float*)d_in[1];
  const long long* edges  = (const long long*)d_in[2];   // int64 per reference
  float*           out    = (float*)d_out;

  char* ws = (char*)d_ws;
  _Float16* Ef16 = (_Float16*)(ws + WS_EF16_OFF);
  float*    En   = (float*)(ws + WS_EN_OFF);
  _Float16* Qf16 = (_Float16*)(ws + WS_QF16_OFF);
  int*      nbr  = (int*)(ws + WS_NBR_OFF);

  const int rows = N_HEADS * N_NODES;                    // 200,000
  MAD_cvt_norm_kernel<<<(rows + 255) / 256, 256, 0, stream>>>(embeds, Ef16, En,
                                                             rows);
  MAD_gatherq_kernel<<<(N_HEADS * N_QUERY + 255) / 256, 256, 0, stream>>>(
      embeds, edges, Qf16);
  MAD_knn_kernel<<<N_HEADS * (N_QUERY / 16), 256, 0, stream>>>(Ef16, En, Qf16,
                                                               edges, nbr);
  MAD_final_kernel<<<(N_BATCH + 255) / 256, 256, 0, stream>>>(embeds, field,
                                                              edges, nbr, out);
}
